// EdgeAttentionAggregator_68624987455535
// MI455X (gfx1250) — matrix operand order repaired and checked by reading, not compiled, not run on hardware
//
#include <hip/hip_runtime.h>
#include <cmath>

// ---------------------------------------------------------------------------
// EdgeAttentionAggregator, restructured for MI455X (gfx1250, wave32, WMMA).
//
// scores = E_nb @ (Wk^T q) + q.bk   ;  context = Wv @ (attn @ E_nb) + bv
// => only four batched GEMMs (all WMMA f16->f32) + one HBM-streaming
//    softmax/aggregate kernel (536MB of neighbor_edges, ~23us roofline),
//    which now stages its LDS tile with GLOBAL_LOAD_ASYNC_TO_LDS_B128.
// ---------------------------------------------------------------------------

typedef _Float16 half_t;
typedef __attribute__((ext_vector_type(16))) _Float16 v16h;
typedef __attribute__((ext_vector_type(8)))  float    v8f;

#define NB        16384     // B
#define KNBR      64
#define NODE_DIM  256
#define EDGE_DIM  128
#define HID       256
#define CTX_DIM   640       // 2*NODE_DIM + EDGE_DIM
#define INV_SQRT_HD 0.17677669529663687f   // 1/sqrt(32)

// ---- WMMA fragment helpers (k contiguous per lane half, wave32) -------------------

__device__ __forceinline__ v8f wmma16(v16h a, v16h b, v8f c) {
  return __builtin_amdgcn_wmma_f32_16x16x32_f16(
      /*neg_a=*/false, a, /*neg_b=*/false, b,
      /*c_mod=*/(short)0, c, /*reuse_a=*/false, /*reuse_b=*/false);
}

// A tile 16x32 (f16) from LDS; base points at A[0][k0], row_stride in halfs.
// lane L: m = L%16, k = (L/16)*16 + h
__device__ __forceinline__ v16h load_a_frag(const half_t* base, int lane,
                                            int row_stride) {
  const half_t* p = base + (lane & 15) * row_stride + (lane >> 4) * 16;
  v16h a;
#pragma unroll
  for (int h = 0; h < 8; ++h) a[h] = p[h];
#pragma unroll
  for (int h = 0; h < 8; ++h) a[8 + h] = p[8 + h];
  return a;
}

// B tile 32x16: B[kk][n] = W[(n0+n)*ldw + k0+kk]  (W row-major [N,K])
// lane L: n = L%16, kk = (L/16)*16 + h  -> 16 contiguous f32 per lane
__device__ __forceinline__ v16h load_b_rowmajor(const float* W, int ldw,
                                                int n0, int k0, int lane) {
  const float* p = W + (size_t)(n0 + (lane & 15)) * ldw + k0 + (lane >> 4) * 16;
  v16h b;
#pragma unroll
  for (int h = 0; h < 16; ++h) b[h] = (half_t)p[h];
  return b;
}

// B tile 32x16: B[kk][n] = W[(k0+kk)*ldw + n0+n]  (W row-major [K,N])
__device__ __forceinline__ v16h load_b_colmajor(const float* W, int ldw,
                                                int n0, int k0, int lane) {
  const float* p = W + (size_t)(k0 + (lane >> 4) * 16) * ldw + n0 + (lane & 15);
  v16h b;
#pragma unroll
  for (int h = 0; h < 16; ++h) b[h] = (half_t)p[h * ldw];
  return b;
}

// ---------------------------------------------------------------------------
// Kernel 1: Q = ctx @ Wq^T + bq ; qk = Q @ Wk ; qb = Q . bk
// 32 edges / block, 8 waves. WMMA throughout.
// ---------------------------------------------------------------------------
__global__ __launch_bounds__(256) void k1_query_project(
    const float* __restrict__ h_u, const float* __restrict__ h_v,
    const float* __restrict__ x_e,
    const float* __restrict__ Wq, const float* __restrict__ bq,
    const float* __restrict__ Wk, const float* __restrict__ bk,
    float* __restrict__ qk_out, float* __restrict__ qb_out) {
  __shared__ half_t ctx[32 * CTX_DIM];  // 40 KB
  __shared__ half_t qh[32 * HID];       // 16 KB
  const int tid = threadIdx.x;
  const int lane = tid & 31, wave = tid >> 5;
  const int e0 = blockIdx.x * 32;

  // Stage concat(h_u, h_v, x_e) as f16
  for (int idx = tid; idx < 32 * CTX_DIM; idx += 256) {
    int m = idx / CTX_DIM, c = idx - m * CTX_DIM;
    int row = e0 + m;
    float v;
    if (c < NODE_DIM)            v = h_u[(size_t)row * NODE_DIM + c];
    else if (c < 2 * NODE_DIM)   v = h_v[(size_t)row * NODE_DIM + c - NODE_DIM];
    else                         v = x_e[(size_t)row * EDGE_DIM + c - 2 * NODE_DIM];
    ctx[idx] = (half_t)v;
  }
  __syncthreads();

  // Q[32,256] = ctx[32,640] @ Wq^T : wave owns 2 N-tiles x 2 M-tiles
  {
    const int n0 = wave * 32;
    v8f acc[2][2] = {};
    for (int k0 = 0; k0 < CTX_DIM; k0 += 32) {
      v16h b0 = load_b_rowmajor(Wq, CTX_DIM, n0, k0, lane);
      v16h b1 = load_b_rowmajor(Wq, CTX_DIM, n0 + 16, k0, lane);
#pragma unroll
      for (int mt = 0; mt < 2; ++mt) {
        v16h a = load_a_frag(&ctx[mt * 16 * CTX_DIM + k0], lane, CTX_DIM);
        acc[mt][0] = wmma16(a, b0, acc[mt][0]);
        acc[mt][1] = wmma16(a, b1, acc[mt][1]);
      }
    }
#pragma unroll
    for (int mt = 0; mt < 2; ++mt)
#pragma unroll
      for (int nt = 0; nt < 2; ++nt) {
        int n = n0 + nt * 16 + (lane & 15);
#pragma unroll
        for (int r = 0; r < 8; ++r) {
          int m = mt * 16 + (lane >> 4) * 8 + r;
          qh[m * HID + n] = (half_t)(acc[mt][nt][r] + bq[n]);
        }
      }
  }
  __syncthreads();

  // qk[32,128] = Q[32,256] @ Wk[256,128] : wave owns 1 N-tile x 2 M-tiles
  {
    const int n0 = wave * 16;
    v8f acc[2] = {};
    for (int k0 = 0; k0 < HID; k0 += 32) {
      v16h b = load_b_colmajor(Wk, EDGE_DIM, n0, k0, lane);
#pragma unroll
      for (int mt = 0; mt < 2; ++mt) {
        v16h a = load_a_frag(&qh[mt * 16 * HID + k0], lane, HID);
        acc[mt] = wmma16(a, b, acc[mt]);
      }
    }
#pragma unroll
    for (int mt = 0; mt < 2; ++mt) {
      int n = n0 + (lane & 15);
#pragma unroll
      for (int r = 0; r < 8; ++r) {
        int m = mt * 16 + (lane >> 4) * 8 + r;
        qk_out[(size_t)(e0 + m) * EDGE_DIM + n] = acc[mt][r];
      }
    }
  }

  // qb = Q . bk (per edge)
  if (tid < 32) {
    float s = 0.f;
    for (int h = 0; h < HID; ++h) s += (float)qh[tid * HID + h] * bk[h];
    qb_out[e0 + tid] = s;
  }
}

// ---------------------------------------------------------------------------
// Kernel 2: masked softmax over 64 neighbors + weighted aggregate of edges.
// One edge per 64-thread block. The 32KB neighbor tile is staged into LDS
// with GLOBAL_LOAD_ASYNC_TO_LDS_B128 (ASYNCcnt path, no VGPR round-trip),
// padded row stride 129 -> conflict-free row AND column access afterwards.
// This kernel is the HBM roofline piece (~536MB streamed once).
// ---------------------------------------------------------------------------
__global__ __launch_bounds__(64) void k2_softmax_agg(
    const float* __restrict__ nbr, const unsigned char* __restrict__ mask,
    const float* __restrict__ qk, const float* __restrict__ qb,
    float* __restrict__ agg) {
  __shared__ float E[KNBR * 129];  // padded stride 129
  __shared__ float qv[EDGE_DIM];
  __shared__ float sc[KNBR];
  const int t = threadIdx.x;
  const int b = blockIdx.x;

  qv[t]      = qk[(size_t)b * EDGE_DIM + t];
  qv[t + 64] = qk[(size_t)b * EDGE_DIM + t + 64];

  // ---- async DMA of the 64x128 f32 tile into LDS (16B per lane per issue)
  {
    const float* gbase = nbr + (size_t)b * KNBR * EDGE_DIM;  // block-uniform
    // low 32 bits of the generic pointer == LDS byte offset
    unsigned lds_base = (unsigned)(unsigned long long)(uintptr_t)&E[0];
#pragma unroll 4
    for (int i = 0; i < 32; ++i) {
      int idx4 = i * 64 + t;          // float4 index within the tile
      int f = idx4 * 4;               // flat float index
      unsigned lds_addr =
          lds_base + (unsigned)(((f >> 7) * 129 + (f & 127)) * 4);
      unsigned goff = (unsigned)idx4 * 16u;  // byte offset from gbase
      asm volatile("global_load_async_to_lds_b128 %0, %1, %2"
                   :
                   : "v"(lds_addr), "v"(goff), "s"(gbase)
                   : "memory");
    }
    asm volatile("s_wait_asynccnt 0x0" ::: "memory");
  }
  __syncthreads();

  // score for neighbor t
  float s = 0.f;
  for (int d = 0; d < EDGE_DIM; ++d) s += E[t * 129 + d] * qv[d];
  s = (s + qb[b]) * INV_SQRT_HD;
  sc[t] = (mask[(size_t)b * KNBR + t] != 0) ? s : -INFINITY;
  __syncthreads();

  float mx = -INFINITY;
  for (int j = 0; j < KNBR; ++j) mx = fmaxf(mx, sc[j]);
  float sum = 0.f;
  for (int j = 0; j < KNBR; ++j) sum += __expf(sc[j] - mx);
  float attn = __expf(sc[t] - mx) / sum;   // masked -> exp(-inf)=0
  __syncthreads();
  sc[t] = attn;
  __syncthreads();

  float a0 = 0.f, a1 = 0.f;
  for (int j = 0; j < KNBR; ++j) {
    float w = sc[j];
    a0 += w * E[j * 129 + t];
    a1 += w * E[j * 129 + t + 64];
  }
  agg[(size_t)b * EDGE_DIM + t]      = a0;
  agg[(size_t)b * EDGE_DIM + t + 64] = a1;
}

// ---------------------------------------------------------------------------
// Kernel 3: context = agg @ Wv^T + bv ; pre = context @ Wo^T + bo + x_e ;
// LayerNorm(pre) * gamma + beta. 32 edges / block, WMMA for both GEMMs.
// ---------------------------------------------------------------------------
__global__ __launch_bounds__(256) void k3_out_project_ln(
    const float* __restrict__ agg, const float* __restrict__ x_e,
    const float* __restrict__ Wv, const float* __restrict__ bv,
    const float* __restrict__ Wo, const float* __restrict__ bo,
    const float* __restrict__ gamma, const float* __restrict__ beta,
    float* __restrict__ out) {
  __shared__ half_t aggh[32 * EDGE_DIM];  // 8 KB
  __shared__ half_t ctxh[32 * HID];       // 16 KB
  __shared__ float  yv[32 * EDGE_DIM];    // 16 KB
  __shared__ float  mu_s[32], rs_s[32];
  const int tid = threadIdx.x;
  const int lane = tid & 31, wave = tid >> 5;
  const int e0 = blockIdx.x * 32;

  for (int idx = tid; idx < 32 * EDGE_DIM; idx += 256)
    aggh[idx] = (half_t)agg[(size_t)e0 * EDGE_DIM + idx];
  __syncthreads();

  // context[32,256] = agg[32,128] @ Wv^T : 2 N-tiles x 2 M-tiles per wave
  {
    const int n0 = wave * 32;
    v8f acc[2][2] = {};
    for (int k0 = 0; k0 < EDGE_DIM; k0 += 32) {
      v16h b0 = load_b_rowmajor(Wv, EDGE_DIM, n0, k0, lane);
      v16h b1 = load_b_rowmajor(Wv, EDGE_DIM, n0 + 16, k0, lane);
#pragma unroll
      for (int mt = 0; mt < 2; ++mt) {
        v16h a = load_a_frag(&aggh[mt * 16 * EDGE_DIM + k0], lane, EDGE_DIM);
        acc[mt][0] = wmma16(a, b0, acc[mt][0]);
        acc[mt][1] = wmma16(a, b1, acc[mt][1]);
      }
    }
#pragma unroll
    for (int mt = 0; mt < 2; ++mt)
#pragma unroll
      for (int nt = 0; nt < 2; ++nt) {
        int n = n0 + nt * 16 + (lane & 15);
#pragma unroll
        for (int r = 0; r < 8; ++r) {
          int m = mt * 16 + (lane >> 4) * 8 + r;
          ctxh[m * HID + n] = (half_t)(acc[mt][nt][r] + bv[n]);
        }
      }
  }
  __syncthreads();

  // y[32,128] = context @ Wo^T + bo + x_e : 1 N-tile x 2 M-tiles per wave
  {
    const int n0 = wave * 16;
    v8f acc[2] = {};
    for (int k0 = 0; k0 < HID; k0 += 32) {
      v16h b = load_b_rowmajor(Wo, HID, n0, k0, lane);
#pragma unroll
      for (int mt = 0; mt < 2; ++mt) {
        v16h a = load_a_frag(&ctxh[mt * 16 * HID + k0], lane, HID);
        acc[mt] = wmma16(a, b, acc[mt]);
      }
    }
#pragma unroll
    for (int mt = 0; mt < 2; ++mt) {
      int n = n0 + (lane & 15);
#pragma unroll
      for (int r = 0; r < 8; ++r) {
        int m = mt * 16 + (lane >> 4) * 8 + r;
        yv[m * EDGE_DIM + n] =
            acc[mt][r] + bo[n] + x_e[(size_t)(e0 + m) * EDGE_DIM + n];
      }
    }
  }
  __syncthreads();

  if (tid < 32) {
    float s = 0.f;
    for (int c = 0; c < EDGE_DIM; ++c) s += yv[tid * EDGE_DIM + c];
    float mu = s * (1.0f / EDGE_DIM);
    float v = 0.f;
    for (int c = 0; c < EDGE_DIM; ++c) {
      float d = yv[tid * EDGE_DIM + c] - mu;
      v += d * d;
    }
    mu_s[tid] = mu;
    rs_s[tid] = rsqrtf(v * (1.0f / EDGE_DIM) + 1e-5f);
  }
  __syncthreads();

  for (int idx = tid; idx < 32 * EDGE_DIM; idx += 256) {
    int m = idx >> 7, c = idx & 127;
    out[(size_t)e0 * EDGE_DIM + idx] =
        (yv[idx] - mu_s[m]) * rs_s[m] * gamma[c] + beta[c];
  }
}

// ---------------------------------------------------------------------------
extern "C" void kernel_launch(void* const* d_in, const int* in_sizes, int n_in,
                              void* d_out, int out_size, void* d_ws,
                              size_t ws_size, hipStream_t stream) {
  const float* h_u  = (const float*)d_in[0];
  const float* h_v  = (const float*)d_in[1];
  const float* x_e  = (const float*)d_in[2];
  const float* nbr  = (const float*)d_in[3];
  const unsigned char* mask = (const unsigned char*)d_in[4];
  const float* Wq = (const float*)d_in[5];
  const float* bq = (const float*)d_in[6];
  const float* Wk = (const float*)d_in[7];
  const float* bk = (const float*)d_in[8];
  const float* Wv = (const float*)d_in[9];
  const float* bv = (const float*)d_in[10];
  const float* Wo = (const float*)d_in[11];
  const float* bo = (const float*)d_in[12];
  const float* gamma = (const float*)d_in[13];
  const float* beta  = (const float*)d_in[14];
  float* out = (float*)d_out;

  float* ws  = (float*)d_ws;
  float* qk  = ws;                                 // NB*128
  float* qb  = ws + (size_t)NB * EDGE_DIM;         // NB
  float* agg = qb + NB;                            // NB*128

  k1_query_project<<<NB / 32, 256, 0, stream>>>(h_u, h_v, x_e, Wq, bq, Wk, bk,
                                                qk, qb);
  k2_softmax_agg<<<NB, 64, 0, stream>>>(nbr, mask, qk, qb, agg);
  k3_out_project_ln<<<NB / 32, 256, 0, stream>>>(agg, x_e, Wv, bv, Wo, bo,
                                                 gamma, beta, out);
}
